// BankedDenoiser_55310588838264
// MI455X (gfx1250) — compile-verified, matrix operand order checked by reading, not run on hardware
//
#include <hip/hip_runtime.h>
#include <cmath>
#include <cstdint>

// ---------------------------------------------------------------------------
// Problem constants (from reference)
// ---------------------------------------------------------------------------
constexpr int B_   = 8;
constexpr int S_   = 1024;
constexpr int IN_  = 256;
constexpr int D_   = 512;
constexpr int H_   = 8;
constexpr int L_   = 4;
constexpr int M_   = 256;
constexpr int TOPK_= 4;
constexpr int DFF_ = 2048;
constexpr int DH_  = D_ / H_;        // 64
constexpr int BS_  = B_ * S_;        // 8192
constexpr int BM_  = B_ * M_;        // 2048

constexpr float TAU_   = 1.0f;
constexpr float GAMMA_ = 0.3f;
constexpr float BETA_  = 1.0f;
constexpr float ETA_   = 1.0f;

// Use CDNA5 async global->LDS copies (ASYNCcnt) for GEMM staging.
#define USE_ASYNC 1

typedef __attribute__((ext_vector_type(16))) __bf16 v16bf;
typedef __attribute__((ext_vector_type(8)))  float  v8f;

// fp32 -> bf16 round-to-nearest-even, pure bit math
__device__ __forceinline__ __bf16 f2bf(float f) {
  unsigned int u = __float_as_uint(f);
  unsigned int r = (u + 0x7FFFu + ((u >> 16) & 1u)) >> 16;
  unsigned short s = (unsigned short)r;
  return __builtin_bit_cast(__bf16, s);
}

__device__ __forceinline__ v8f wmma_bf16(v16bf a, v16bf b, v8f c) {
  return __builtin_amdgcn_wmma_f32_16x16x32_bf16(false, a, false, b,
                                                 (short)0, c, false, false);
}

#if USE_ASYNC
// Flat LDS addresses carry the LDS byte offset in their low 32 bits (ISA §10.2).
__device__ __forceinline__ unsigned lds_off(const void* p) {
  return (unsigned)(unsigned long long)p;
}
// 16B per lane, ASYNCcnt-tracked, data goes straight to LDS (no VGPR bounce).
__device__ __forceinline__ void async_b128(unsigned lds_byte, const void* gaddr) {
  asm volatile("global_load_async_to_lds_b128 %0, %1, off"
               :: "v"(lds_byte), "v"(gaddr) : "memory");
}
__device__ __forceinline__ void wait_async0() {
  asm volatile("s_wait_asynccnt 0" ::: "memory");
}
#endif

// ---------------------------------------------------------------------------
// WMMA bf16 GEMM (B always [N][K], K-contiguous):
//   C[z] = alpha * A[z] @ B[z]^T (+bias) (optional relu)
// Block tile 128(M) x 64(N) x 32(K); 256 threads = 8 wave32s in a 4x2 grid;
// each wave owns a 32x32 output (2x2 WMMA accumulators). Double-buffered LDS
// staged with global_load_async_to_lds_b128.
// Batch z decomposes as (outer, head) with independent strides.
// ---------------------------------------------------------------------------
#define TILE_M 128
#define TILE_N 64
#define TILE_K 32
#define LDS_PITCH 40   // bf16 elems per row: 80B, keeps 16B alignment

template <bool RELU>
__global__ __launch_bounds__(256) void gemm_wmma(
    const __bf16* __restrict__ A, const __bf16* __restrict__ Bm,
    float* __restrict__ C, const float* __restrict__ bias,
    int Kdim, int lda, int ldb, int ldc,
    long long sAb, long long sAh, long long sBb, long long sBh,
    long long sCb, long long sCh, int Hbat, float alpha) {
  __shared__ __bf16 sA[2][TILE_M][LDS_PITCH];
  __shared__ __bf16 sB[2][TILE_N][LDS_PITCH];

  const int tid  = threadIdx.x;
  const int lane = tid & 31;
  const int w    = tid >> 5;
  const int wm   = w >> 1;   // 0..3 : 32-row slice
  const int wn   = w & 1;    // 0..1 : 32-col slice

  const int bz = blockIdx.z;
  const int bb = bz / Hbat;
  const int hh = bz % Hbat;
  const __bf16* Ab = A  + bb * sAb + hh * sAh;
  const __bf16* Bb = Bm + bb * sBb + hh * sBh;
  float*        Cb = C  + bb * sCb + hh * sCh;

  const int row0 = blockIdx.y * TILE_M;
  const int col0 = blockIdx.x * TILE_N;

  // Staging geometry: each thread moves three 16B chunks per k-step.
  const int tr   = tid >> 2;         // 0..63
  const int tseg = (tid & 3) * 8;    // bf16 offset of 16B chunk
  const __bf16* aSrc0 = Ab + (long long)(row0 + tr)      * lda + tseg;
  const __bf16* aSrc1 = Ab + (long long)(row0 + 64 + tr) * lda + tseg;
  const __bf16* bSrc  = Bb + (long long)(col0 + tr)      * ldb + tseg;

  auto stage = [&](int buf, int k0) {
#if USE_ASYNC
    async_b128(lds_off(&sA[buf][tr][tseg]),      aSrc0 + k0);
    async_b128(lds_off(&sA[buf][64 + tr][tseg]), aSrc1 + k0);
    async_b128(lds_off(&sB[buf][tr][tseg]),      bSrc  + k0);
#else
#pragma unroll
    for (int i = 0; i < 8; ++i) {
      sA[buf][tr][tseg + i]      = aSrc0[k0 + i];
      sA[buf][64 + tr][tseg + i] = aSrc1[k0 + i];
      sB[buf][tr][tseg + i]      = bSrc[k0 + i];
    }
#endif
  };

  v8f c00 = {0,0,0,0,0,0,0,0}, c01 = c00, c10 = c00, c11 = c00;
  const int lr  = lane & 15;
  const int kb_ = (lane >> 4) * 8;

  stage(0, 0);
#if USE_ASYNC
  wait_async0();
#endif
  __syncthreads();

  for (int k0 = 0; k0 < Kdim; k0 += TILE_K) {
    const int  cur  = (k0 >> 5) & 1;
    const bool more = (k0 + TILE_K) < Kdim;
    if (more) stage(cur ^ 1, k0 + TILE_K);   // overlaps with WMMAs below

    // Fragment layout per ISA: lane<16 -> K {kb..kb+7, kb+16..kb+23}
    v16bf a0, a1, b0, b1;
#pragma unroll
    for (int i = 0; i < 16; ++i) {
      const int kk = kb_ + (i < 8 ? i : i + 8);
      a0[i] = sA[cur][wm * 32 + lr][kk];
      a1[i] = sA[cur][wm * 32 + 16 + lr][kk];
      b0[i] = sB[cur][wn * 32 + lr][kk];
      b1[i] = sB[cur][wn * 32 + 16 + lr][kk];
    }
    c00 = wmma_bf16(a0, b0, c00);
    c01 = wmma_bf16(a0, b1, c01);
    c10 = wmma_bf16(a1, b0, c10);
    c11 = wmma_bf16(a1, b1, c11);

#if USE_ASYNC
    if (more) wait_async0();
#endif
    __syncthreads();
  }

  // C/D layout: col = lane&15 ; row = (lane>>4)*8 + r
  auto store_tile = [&](v8f& acc, int sm, int sn) {
    const int cn = col0 + wn * 32 + sn * 16 + (lane & 15);
    const int rb = row0 + wm * 32 + sm * 16 + (lane >> 4) * 8;
    const float bv = bias ? bias[cn] : 0.0f;
#pragma unroll
    for (int r = 0; r < 8; ++r) {
      float v = acc[r] * alpha + bv;
      if (RELU) v = fmaxf(v, 0.0f);
      Cb[(long long)(rb + r) * ldc + cn] = v;
    }
  };
  store_tile(c00, 0, 0); store_tile(c01, 0, 1);
  store_tile(c10, 1, 0); store_tile(c11, 1, 1);
}

// ---------------------------------------------------------------------------
// Data-layout kernels
// ---------------------------------------------------------------------------
__global__ void f32_to_bf16(const float* __restrict__ in, __bf16* __restrict__ out,
                            long long n) {
  long long i = (long long)blockIdx.x * blockDim.x + threadIdx.x;
  const long long stride = (long long)gridDim.x * blockDim.x;
  for (; i < n; i += stride) out[i] = f2bf(in[i]);
}

// W [R][C] fp32  ->  out [C][R] bf16 (LDS-tiled transpose; R,C multiples of 32)
__global__ __launch_bounds__(256) void transpose_f32_to_bf16(
    const float* __restrict__ in, __bf16* __restrict__ out, int R, int C) {
  __shared__ __bf16 tile[32][33];
  const int bx = blockIdx.x * 32;   // along C
  const int by = blockIdx.y * 32;   // along R
  const int tx = threadIdx.x & 31;
  const int ty = threadIdx.x >> 5;  // 0..7
#pragma unroll
  for (int i = 0; i < 32; i += 8)
    tile[ty + i][tx] = f2bf(in[(long long)(by + ty + i) * C + bx + tx]);
  __syncthreads();
#pragma unroll
  for (int i = 0; i < 32; i += 8)
    out[(long long)(bx + ty + i) * R + by + tx] = tile[tx][ty + i];
}

// qkv fp32 [B,S,3D] -> Q,K bf16 [B,H,S,DH] and V^T bf16 [B,H,DH,S]
__global__ void split_qkv(const float* __restrict__ qkv, __bf16* __restrict__ qb,
                          __bf16* __restrict__ kb, __bf16* __restrict__ vt) {
  const long long n = (long long)BS_ * D_;
  long long i = (long long)blockIdx.x * blockDim.x + threadIdx.x;
  const long long stride = (long long)gridDim.x * blockDim.x;
  for (; i < n; i += stride) {
    const int dd = (int)(i % D_);
    const long long bs = i / D_;
    const int s = (int)(bs % S_);
    const int b = (int)(bs / S_);
    const int hh = dd / DH_, d = dd % DH_;
    const float* src = qkv + bs * (long long)(3 * D_);
    const long long qi = ((long long)(b * H_ + hh) * S_ + s) * DH_ + d;
    qb[qi] = f2bf(src[dd]);
    kb[qi] = f2bf(src[D_ + dd]);
    vt[((long long)(b * H_ + hh) * DH_ + d) * S_ + s] = f2bf(src[2 * D_ + dd]);
  }
}

// bank V fp32 [B*M, D] -> V^T bf16 [B,H,DH,M]
__global__ void bank_vt(const float* __restrict__ bv, __bf16* __restrict__ vt) {
  const long long n = (long long)BM_ * D_;
  long long i = (long long)blockIdx.x * blockDim.x + threadIdx.x;
  const long long stride = (long long)gridDim.x * blockDim.x;
  for (; i < n; i += stride) {
    const int dd = (int)(i % D_);
    const long long bm = i / D_;
    const int m = (int)(bm % M_);
    const int b = (int)(bm / M_);
    const int hh = dd / DH_, d = dd % DH_;
    vt[((long long)(b * H_ + hh) * DH_ + d) * M_ + m] = f2bf(bv[i]);
  }
}

// ---------------------------------------------------------------------------
// Elementwise / reduction kernels
// ---------------------------------------------------------------------------
__global__ void add_pe_temb(float* __restrict__ h, const float* __restrict__ temb) {
  const long long n = (long long)BS_ * D_;
  long long i = (long long)blockIdx.x * blockDim.x + threadIdx.x;
  const long long stride = (long long)gridDim.x * blockDim.x;
  const float c = -logf(10000.0f) / (float)D_;
  for (; i < n; i += stride) {
    const int d = (int)(i % D_);
    const long long t = i / D_;
    const int s = (int)(t % S_);
    const int b = (int)(t / S_);
    const float freq = expf((float)(2 * (d >> 1)) * c);
    const float ang = (float)s * freq;
    const float pe = (d & 1) ? cosf(ang) : sinf(ang);
    h[i] += pe + temb[b * D_ + d];
  }
}

__global__ __launch_bounds__(256) void softmax_rows_bf16(
    const float* __restrict__ in, __bf16* __restrict__ out, int ncols) {
  __shared__ float red[256];
  const long long row = blockIdx.x;
  const float* src = in + row * (long long)ncols;
  __bf16* dst = out + row * (long long)ncols;
  const int tid = threadIdx.x;

  float mx = -3.4e38f;
  for (int c = tid; c < ncols; c += 256) mx = fmaxf(mx, src[c]);
  red[tid] = mx; __syncthreads();
  for (int s = 128; s; s >>= 1) { if (tid < s) red[tid] = fmaxf(red[tid], red[tid + s]); __syncthreads(); }
  mx = red[0]; __syncthreads();

  float sum = 0.f;
  for (int c = tid; c < ncols; c += 256) sum += expf(src[c] - mx);
  red[tid] = sum; __syncthreads();
  for (int s = 128; s; s >>= 1) { if (tid < s) red[tid] += red[tid + s]; __syncthreads(); }
  const float inv = 1.0f / red[0];
  for (int c = tid; c < ncols; c += 256) dst[c] = f2bf(expf(src[c] - mx) * inv);
}

// h = LN(h + add) * g + b   (row-wise over ncols)
__global__ __launch_bounds__(256) void residual_ln(
    float* __restrict__ h, const float* __restrict__ add,
    const float* __restrict__ g, const float* __restrict__ bvec, int ncols) {
  __shared__ float red[256];
  __shared__ float s_mu, s_rs;
  const long long row = blockIdx.x;
  float* hp = h + row * (long long)ncols;
  const float* ap = add + row * (long long)ncols;
  const int tid = threadIdx.x;

  float acc = 0.f;
  for (int c = tid; c < ncols; c += 256) acc += hp[c] + ap[c];
  red[tid] = acc; __syncthreads();
  for (int s = 128; s; s >>= 1) { if (tid < s) red[tid] += red[tid + s]; __syncthreads(); }
  if (tid == 0) s_mu = red[0] / (float)ncols;
  __syncthreads();
  const float mu = s_mu;

  acc = 0.f;
  for (int c = tid; c < ncols; c += 256) { float v = hp[c] + ap[c] - mu; acc += v * v; }
  __syncthreads();
  red[tid] = acc; __syncthreads();
  for (int s = 128; s; s >>= 1) { if (tid < s) red[tid] += red[tid + s]; __syncthreads(); }
  if (tid == 0) s_rs = rsqrtf(red[0] / (float)ncols + 1e-5f);
  __syncthreads();
  const float rs = s_rs;

  for (int c = tid; c < ncols; c += 256)
    hp[c] = (hp[c] + ap[c] - mu) * rs * g[c] + bvec[c];
}

// per (b,m): p2 = sum Phi^2 ; keyb = GAMMA*log(Size+1e-6) - 0.5*mean(Sig^2)
__global__ __launch_bounds__(256) void bank_stats(
    const float* __restrict__ Phi, const float* __restrict__ Sig,
    const float* __restrict__ Size, float* __restrict__ p2, float* __restrict__ keyb) {
  __shared__ float r1[256], r2[256];
  const int row = blockIdx.x;   // b*M + m
  const float* ph = Phi + (long long)row * D_;
  const float* sg = Sig + (long long)row * D_;
  const int tid = threadIdx.x;
  float a = 0.f, c = 0.f;
  for (int d = tid; d < D_; d += 256) {
    const float pv = ph[d]; a += pv * pv;
    const float sv = sg[d]; c += sv * sv;
  }
  r1[tid] = a; r2[tid] = c; __syncthreads();
  for (int s = 128; s; s >>= 1) {
    if (tid < s) { r1[tid] += r1[tid + s]; r2[tid] += r2[tid + s]; }
    __syncthreads();
  }
  if (tid == 0) {
    p2[row] = r1[0];
    const float sig = r2[0] / (float)D_;
    keyb[row] = GAMMA_ * logf(Size[row] + 1e-6f) - 0.5f * sig;
  }
}

// scores = BETA*(ETA*dot - dist2/(TAU*D)) + keyb[n] ; masked -> -1e9  (in-place)
__global__ void bank_combine(float* __restrict__ dot, const float* __restrict__ phid,
                             const float* __restrict__ p2, const float* __restrict__ keyb,
                             const unsigned char* __restrict__ mask) {
  const long long n = (long long)B_ * H_ * M_ * M_;
  long long i = (long long)blockIdx.x * blockDim.x + threadIdx.x;
  const long long stride = (long long)gridDim.x * blockDim.x;
  for (; i < n; i += stride) {
    const int nn = (int)(i % M_);
    const long long t = i / M_;
    const int mm = (int)(t % M_);
    const int b = (int)((t / M_) / H_);
    const float dist2 = p2[b * M_ + mm] + p2[b * M_ + nn]
                      - 2.0f * phid[((long long)b * M_ + mm) * M_ + nn];
    float s = BETA_ * (ETA_ * dot[i] - dist2 * (1.0f / (TAU_ * (float)D_)))
            + keyb[b * M_ + nn];
    if (!mask[b * M_ + nn]) s = -1e9f;
    dot[i] = s;
  }
}

// per token (b,s): top-4 over 256 logits, softmax gates, mix Z rows into h
__global__ __launch_bounds__(256) void route_topk_mix(
    const float* __restrict__ logits, const unsigned char* __restrict__ mask,
    const float* __restrict__ h, const float* __restrict__ Z,
    float* __restrict__ routed, __bf16* __restrict__ routedb) {
  __shared__ float cur[256];
  __shared__ float sv[256];
  __shared__ int   si[256];
  __shared__ float wsh[TOPK_];
  __shared__ int   ish[TOPK_];
  const long long row = blockIdx.x;          // b*S + s
  const int b = (int)(row / S_);
  const int tid = threadIdx.x;               // == m (M_ == 256)

  float v = logits[row * M_ + tid];
  if (!mask[b * M_ + tid]) v = -1e9f;
  cur[tid] = v; __syncthreads();

  for (int k = 0; k < TOPK_; ++k) {
    sv[tid] = cur[tid]; si[tid] = tid; __syncthreads();
    for (int s = 128; s; s >>= 1) {
      if (tid < s && sv[tid + s] > sv[tid]) { sv[tid] = sv[tid + s]; si[tid] = si[tid + s]; }
      __syncthreads();
    }
    if (tid == 0) { ish[k] = si[0]; wsh[k] = sv[0]; cur[si[0]] = -3.4e38f; }
    __syncthreads();
  }
  if (tid == 0) {
    const float mx = wsh[0];
    float s = 0.f;
    for (int k = 0; k < TOPK_; ++k) { wsh[k] = expf(wsh[k] - mx); s += wsh[k]; }
    for (int k = 0; k < TOPK_; ++k) wsh[k] /= s;
  }
  __syncthreads();

  for (int d = tid; d < D_; d += 256) {
    float acc = h[row * D_ + d];
#pragma unroll
    for (int k = 0; k < TOPK_; ++k)
      acc += wsh[k] * Z[((long long)b * M_ + ish[k]) * D_ + d];
    routed[row * D_ + d] = acc;
    routedb[row * D_ + d] = f2bf(acc);
  }
}

// ---------------------------------------------------------------------------
// Host-side helpers
// ---------------------------------------------------------------------------
static inline void conv_bf16(hipStream_t s, const float* in, __bf16* out, long long n) {
  long long blk = (n + 255) / 256;
  if (blk > 8192) blk = 8192;
  f32_to_bf16<<<dim3((unsigned)blk), dim3(256), 0, s>>>(in, out, n);
}

static inline void transpose_w(hipStream_t s, const float* in, __bf16* out, int R, int C) {
  transpose_f32_to_bf16<<<dim3(C / 32, R / 32), dim3(256), 0, s>>>(in, out, R, C);
}

static void gemm(hipStream_t s, bool relu,
                 const __bf16* A, const __bf16* Bm, float* C, const float* bias,
                 int M, int N, int K, int lda, int ldb, int ldc,
                 long long sAb, long long sAh, long long sBb, long long sBh,
                 long long sCb, long long sCh, int batch, int Hbat, float alpha) {
  dim3 grid(N / TILE_N, M / TILE_M, batch);
  dim3 block(256);
  if (relu) gemm_wmma<true ><<<grid, block, 0, s>>>(A, Bm, C, bias, K, lda, ldb, ldc, sAb, sAh, sBb, sBh, sCb, sCh, Hbat, alpha);
  else      gemm_wmma<false><<<grid, block, 0, s>>>(A, Bm, C, bias, K, lda, ldb, ldc, sAb, sAh, sBb, sBh, sCb, sCh, Hbat, alpha);
}

// ---------------------------------------------------------------------------
// Forward pipeline
// ---------------------------------------------------------------------------
extern "C" void kernel_launch(void* const* d_in, const int* in_sizes, int n_in,
                              void* d_out, int out_size, void* d_ws, size_t ws_size,
                              hipStream_t stream) {
  (void)in_sizes; (void)n_in; (void)out_size; (void)ws_size;

  const float* x_t      = (const float*)d_in[0];
  const float* t_embed  = (const float*)d_in[1];
  const float* Phi      = (const float*)d_in[2];
  const float* Sig      = (const float*)d_in[3];
  const float* Size     = (const float*)d_in[4];
  const unsigned char* mask = (const unsigned char*)d_in[5];
  const float* Win      = (const float*)d_in[6];
  const float* b_in     = (const float*)d_in[7];
  const float* Wout     = (const float*)d_in[8];
  const float* b_out    = (const float*)d_in[9];
  const float* enc_Wqkv = (const float*)d_in[10];
  const float* enc_bqkv = (const float*)d_in[11];
  const float* enc_Wo   = (const float*)d_in[12];
  const float* enc_bo   = (const float*)d_in[13];
  const float* ln1_g    = (const float*)d_in[14];
  const float* ln1_b    = (const float*)d_in[15];
  const float* ln2_g    = (const float*)d_in[16];
  const float* ln2_b    = (const float*)d_in[17];
  const float* ff_W1    = (const float*)d_in[18];
  const float* ff_b1    = (const float*)d_in[19];
  const float* ff_W2    = (const float*)d_in[20];
  const float* ff_b2    = (const float*)d_in[21];
  const float* sa_Wq    = (const float*)d_in[22];
  const float* sa_Wk    = (const float*)d_in[23];
  const float* sa_Wv    = (const float*)d_in[24];
  const float* sa_Wo    = (const float*)d_in[25];
  const float* rt_Wq    = (const float*)d_in[26];

  // ---- workspace bump allocator ----
  char* ws = (char*)d_ws;
  size_t off = 0;
  auto alloc = [&](size_t bytes) -> void* {
    void* p = ws + off;
    off = (off + bytes + 255) & ~(size_t)255;
    return p;
  };

  float*  h       = (float*)alloc(sizeof(float) * (size_t)BS_ * D_);
  float*  tmp_big = (float*)alloc(sizeof(float) * (size_t)BS_ * DFF_);   // qkv / ff1
  float*  scores  = (float*)alloc(sizeof(float) * (size_t)B_ * H_ * S_ * S_);
  float*  ctx     = (float*)alloc(sizeof(float) * (size_t)BS_ * D_);
  float*  tmpD    = (float*)alloc(sizeof(float) * (size_t)BS_ * D_);     // attn_out / ff2
  float*  bankQKV = (float*)alloc(sizeof(float) * (size_t)3 * BM_ * D_);
  float*  dot     = (float*)alloc(sizeof(float) * (size_t)B_ * H_ * M_ * M_);
  float*  phid    = (float*)alloc(sizeof(float) * (size_t)B_ * M_ * M_);
  float*  p2      = (float*)alloc(sizeof(float) * (size_t)BM_);
  float*  keyb    = (float*)alloc(sizeof(float) * (size_t)BM_);
  float*  Zctx    = (float*)alloc(sizeof(float) * (size_t)BM_ * D_);
  float*  Zfin    = (float*)alloc(sizeof(float) * (size_t)BM_ * D_);
  float*  hq      = (float*)alloc(sizeof(float) * (size_t)BS_ * D_);
  float*  logits  = (float*)alloc(sizeof(float) * (size_t)BS_ * M_);
  float*  routed  = (float*)alloc(sizeof(float) * (size_t)BS_ * D_);

  __bf16* wT      = (__bf16*)alloc(sizeof(__bf16) * (size_t)D_ * DFF_);  // transposed weight scratch
  __bf16* xb      = (__bf16*)alloc(sizeof(__bf16) * (size_t)BS_ * IN_);
  __bf16* hb      = (__bf16*)alloc(sizeof(__bf16) * (size_t)BS_ * D_);
  __bf16* bigb    = (__bf16*)alloc(sizeof(__bf16) * (size_t)BS_ * DFF_); // ff1b
  __bf16* attnQ   = (__bf16*)alloc(sizeof(__bf16) * (size_t)BS_ * D_);   // [B,H,S,DH]
  __bf16* attnK   = (__bf16*)alloc(sizeof(__bf16) * (size_t)BS_ * D_);   // [B,H,S,DH]
  __bf16* attnVt  = (__bf16*)alloc(sizeof(__bf16) * (size_t)BS_ * D_);   // [B,H,DH,S]
  __bf16* scoresb = (__bf16*)alloc(sizeof(__bf16) * (size_t)B_ * H_ * S_ * S_);
  __bf16* ctxb    = (__bf16*)alloc(sizeof(__bf16) * (size_t)BS_ * D_);
  __bf16* bankQKb = (__bf16*)alloc(sizeof(__bf16) * (size_t)2 * BM_ * D_);
  __bf16* bankVt  = (__bf16*)alloc(sizeof(__bf16) * (size_t)BM_ * D_);   // [B,H,DH,M]
  __bf16* phib    = (__bf16*)alloc(sizeof(__bf16) * (size_t)B_ * M_ * D_);
  __bf16* bankAb  = (__bf16*)alloc(sizeof(__bf16) * (size_t)B_ * H_ * M_ * M_);
  __bf16* Zctxb   = (__bf16*)alloc(sizeof(__bf16) * (size_t)BM_ * D_);
  __bf16* hqb     = (__bf16*)alloc(sizeof(__bf16) * (size_t)BS_ * D_);
  __bf16* routedb = (__bf16*)alloc(sizeof(__bf16) * (size_t)BS_ * D_);

  const float attn_scale = 1.0f / sqrtf((float)DH_);
  const float rt_scale   = 1.0f / sqrtf((float)D_);

  // ---- input projection + positional / time embedding ----
  conv_bf16(stream, x_t, xb, (long long)BS_ * IN_);
  transpose_w(stream, Win, wT, IN_, D_);                 // [D, IN]
  gemm(stream, false, xb, wT, h, b_in,
       BS_, D_, IN_, IN_, IN_, D_, 0,0,0,0,0,0, 1, 1, 1.0f);
  add_pe_temb<<<dim3(8192), dim3(256), 0, stream>>>(h, t_embed);

  // ---- transformer encoder stack (post-LN) ----
  for (int l = 0; l < L_; ++l) {
    conv_bf16(stream, h, hb, (long long)BS_ * D_);
    transpose_w(stream, enc_Wqkv + (size_t)l * D_ * 3 * D_, wT, D_, 3 * D_);  // [3D, D]
    gemm(stream, false, hb, wT, tmp_big, enc_bqkv + (size_t)l * 3 * D_,
         BS_, 3 * D_, D_, D_, D_, 3 * D_, 0,0,0,0,0,0, 1, 1, 1.0f);
    split_qkv<<<dim3(8192), dim3(256), 0, stream>>>(tmp_big, attnQ, attnK, attnVt);

    // scores[b,h] = Q @ K^T * scale
    gemm(stream, false, attnQ, attnK, scores, nullptr,
         S_, S_, DH_, DH_, DH_, S_,
         (long long)S_ * DH_, 0, (long long)S_ * DH_, 0,
         (long long)S_ * S_, 0, B_ * H_, 1, attn_scale);
    softmax_rows_bf16<<<dim3(B_ * H_ * S_), dim3(256), 0, stream>>>(scores, scoresb, S_);

    // ctx[b,s, h*DH+d] = A @ V   (V supplied transposed [DH][S])
    gemm(stream, false, scoresb, attnVt, ctx, nullptr,
         S_, DH_, S_, S_, S_, D_,
         (long long)H_ * S_ * S_, (long long)S_ * S_,
         (long long)H_ * DH_ * S_, (long long)DH_ * S_,
         (long long)S_ * D_, DH_, B_ * H_, H_, 1.0f);

    conv_bf16(stream, ctx, ctxb, (long long)BS_ * D_);
    transpose_w(stream, enc_Wo + (size_t)l * D_ * D_, wT, D_, D_);
    gemm(stream, false, ctxb, wT, tmpD, enc_bo + (size_t)l * D_,
         BS_, D_, D_, D_, D_, D_, 0,0,0,0,0,0, 1, 1, 1.0f);
    residual_ln<<<dim3(BS_), dim3(256), 0, stream>>>(h, tmpD, ln1_g + (size_t)l * D_,
                                                     ln1_b + (size_t)l * D_, D_);

    // FFN
    conv_bf16(stream, h, hb, (long long)BS_ * D_);
    transpose_w(stream, ff_W1 + (size_t)l * D_ * DFF_, wT, D_, DFF_);   // [DFF, D]
    gemm(stream, true /*relu*/, hb, wT, tmp_big, ff_b1 + (size_t)l * DFF_,
         BS_, DFF_, D_, D_, D_, DFF_, 0,0,0,0,0,0, 1, 1, 1.0f);
    conv_bf16(stream, tmp_big, bigb, (long long)BS_ * DFF_);
    transpose_w(stream, ff_W2 + (size_t)l * DFF_ * D_, wT, DFF_, D_);   // [D, DFF]
    gemm(stream, false, bigb, wT, tmpD, ff_b2 + (size_t)l * D_,
         BS_, D_, DFF_, DFF_, DFF_, D_, 0,0,0,0,0,0, 1, 1, 1.0f);
    residual_ln<<<dim3(BS_), dim3(256), 0, stream>>>(h, tmpD, ln2_g + (size_t)l * D_,
                                                     ln2_b + (size_t)l * D_, D_);
  }

  // ---- SetBankAttention ----
  conv_bf16(stream, Phi, phib, (long long)B_ * M_ * D_);
  float*  bq  = bankQKV;
  float*  bk  = bankQKV + (size_t)BM_ * D_;
  float*  bv  = bankQKV + (size_t)2 * BM_ * D_;
  __bf16* bqb = bankQKb;
  __bf16* bkb = bankQKb + (size_t)BM_ * D_;

  transpose_w(stream, sa_Wq, wT, D_, D_);
  gemm(stream, false, phib, wT, bq, nullptr, BM_, D_, D_, D_, D_, D_, 0,0,0,0,0,0, 1, 1, 1.0f);
  transpose_w(stream, sa_Wk, wT, D_, D_);
  gemm(stream, false, phib, wT, bk, nullptr, BM_, D_, D_, D_, D_, D_, 0,0,0,0,0,0, 1, 1, 1.0f);
  transpose_w(stream, sa_Wv, wT, D_, D_);
  gemm(stream, false, phib, wT, bv, nullptr, BM_, D_, D_, D_, D_, D_, 0,0,0,0,0,0, 1, 1, 1.0f);
  conv_bf16(stream, bankQKV, bankQKb, (long long)2 * BM_ * D_);   // q and k are contiguous
  bank_vt<<<dim3(4096), dim3(256), 0, stream>>>(bv, bankVt);

  // dot[b,h] = q @ k^T * scale
  gemm(stream, false, bqb, bkb, dot, nullptr,
       M_, M_, DH_, D_, D_, M_,
       (long long)M_ * D_, DH_, (long long)M_ * D_, DH_,
       (long long)H_ * M_ * M_, (long long)M_ * M_, B_ * H_, H_, attn_scale);
  // phid[b] = Phi @ Phi^T
  gemm(stream, false, phib, phib, phid, nullptr,
       M_, M_, D_, D_, D_, M_,
       (long long)M_ * D_, 0, (long long)M_ * D_, 0,
       (long long)M_ * M_, 0, B_, 1, 1.0f);
  bank_stats<<<dim3(BM_), dim3(256), 0, stream>>>(Phi, Sig, Size, p2, keyb);
  bank_combine<<<dim3(8192), dim3(256), 0, stream>>>(dot, phid, p2, keyb, mask);
  softmax_rows_bf16<<<dim3(B_ * H_ * M_), dim3(256), 0, stream>>>(dot, bankAb, M_);

  // Zctx[b,m, h*DH+d] = A @ v  (v transposed [DH][M])
  gemm(stream, false, bankAb, bankVt, Zctx, nullptr,
       M_, DH_, M_, M_, M_, D_,
       (long long)H_ * M_ * M_, (long long)M_ * M_,
       (long long)H_ * DH_ * M_, (long long)DH_ * M_,
       (long long)M_ * D_, DH_, B_ * H_, H_, 1.0f);
  conv_bf16(stream, Zctx, Zctxb, (long long)BM_ * D_);
  transpose_w(stream, sa_Wo, wT, D_, D_);
  gemm(stream, false, Zctxb, wT, Zfin, nullptr,
       BM_, D_, D_, D_, D_, D_, 0,0,0,0,0,0, 1, 1, 1.0f);

  // ---- TokenSetRouter ----
  conv_bf16(stream, h, hb, (long long)BS_ * D_);
  transpose_w(stream, rt_Wq, wT, D_, D_);
  gemm(stream, false, hb, wT, hq, nullptr,
       BS_, D_, D_, D_, D_, D_, 0,0,0,0,0,0, 1, 1, 1.0f);
  conv_bf16(stream, hq, hqb, (long long)BS_ * D_);
  // logits[b] = hq[b] @ Phi[b]^T / sqrt(D)
  gemm(stream, false, hqb, phib, logits, nullptr,
       S_, M_, D_, D_, D_, M_,
       (long long)S_ * D_, 0, (long long)M_ * D_, 0,
       (long long)S_ * M_, 0, B_, 1, rt_scale);
  route_topk_mix<<<dim3(BS_), dim3(256), 0, stream>>>(logits, mask, h, Zfin, routed, routedb);

  // ---- output projection ----
  transpose_w(stream, Wout, wT, D_, IN_);   // [IN, D]
  gemm(stream, false, routedb, wT, (float*)d_out, b_out,
       BS_, IN_, D_, D_, D_, IN_, 0,0,0,0,0,0, 1, 1, 1.0f);
}
